// Attention_65214783422900
// MI455X (gfx1250) — compile-verified, matrix operand order checked by reading, not run on hardware
//
#include <hip/hip_runtime.h>
#include <hip/hip_bf16.h>

typedef unsigned short u16;
typedef __attribute__((ext_vector_type(16))) __bf16    bf16x16;
typedef __attribute__((ext_vector_type(16))) _Float16  f16x16;
typedef __attribute__((ext_vector_type(8)))  float     f32x8;
typedef __attribute__((ext_vector_type(4)))  float     f32x4;

#define BQ 64
#define SQ 257
#define DQ 768
#define HQ 12
#define HDQ 64
#define ROWS (BQ * SQ)          // 16448
#define ADH 512
#define SPAD 288                // 257 keys padded to 9 * 32
#define EPSQ 1e-10f

// ---------------------------------------------------------------- helpers

__device__ __forceinline__ u16 f2bf(float x) {
    union { float f; unsigned u; } v; v.f = x;
    unsigned r = v.u + 0x7FFFu + ((v.u >> 16) & 1u);
    return (u16)(r >> 16);
}

__device__ __forceinline__ f32x8 zero8() {
    f32x8 z = {0.f,0.f,0.f,0.f,0.f,0.f,0.f,0.f};
    return z;
}

// A/B fragment load: elements [0..7] = p[0..7], [8..15] = p[16..23]
// (16-bit operand VGPR layout, ISA 7.12.2: lane half selects kb = 0/8,
//  VGPRs 0..3 hold K = kb..kb+7, VGPRs 4..7 hold K = kb+16..kb+23)
__device__ __forceinline__ bf16x16 ldbf2(const u16* p0, const u16* p1) {
    union { bf16x16 v; f32x4 f[2]; } u;
    u.f[0] = *reinterpret_cast<const f32x4*>(p0);
    u.f[1] = *reinterpret_cast<const f32x4*>(p1);
    return u.v;
}
__device__ __forceinline__ bf16x16 ldbf(const u16* p) { return ldbf2(p, p + 16); }

__device__ __forceinline__ f16x16 ldh(const _Float16* p) {
    union { f16x16 v; f32x4 f[2]; } u;
    u.f[0] = *reinterpret_cast<const f32x4*>(p);
    u.f[1] = *reinterpret_cast<const f32x4*>(p + 16);
    return u.v;
}

__device__ __forceinline__ f32x8 wmma_bf(bf16x16 a, bf16x16 b, f32x8 c) {
    return __builtin_amdgcn_wmma_f32_16x16x32_bf16(false, a, false, b, (short)0, c, false, false);
}
__device__ __forceinline__ f32x8 wmma_h(f16x16 a, f16x16 b, f32x8 c) {
    return __builtin_amdgcn_wmma_f32_16x16x32_f16(false, a, false, b, (short)0, c, false, false);
}

// gfx1250 async global->LDS copy (tracked by ASYNCcnt), 16 bytes per lane.
// lds operand: low 32 bits of a __shared__ pointer == addrspace(3) byte offset.
__device__ __forceinline__ void async_ld_b128(const u16* gsrc, u16* lds_dst) {
    unsigned           ldsoff = (unsigned)(size_t)(void*)lds_dst;
    unsigned long long gaddr  = (unsigned long long)(size_t)(const void*)gsrc;
    asm volatile("global_load_async_to_lds_b128 %0, %1, off"
                 :: "v"(ldsoff), "v"(gaddr) : "memory");
}
__device__ __forceinline__ void wait_async0() {
    asm volatile("s_wait_asynccnt 0x0" ::: "memory");
}

// ---------------------------------------------------------------- prep

__global__ void k_cvt(const float* __restrict__ src, u16* __restrict__ dst, int n) {
    int i = blockIdx.x * blockDim.x + threadIdx.x;
    if (i < n) dst[i] = f2bf(src[i]);
}

// dst[c*R + r] = src[r*C + c]  (transpose + convert to bf16)
__global__ void k_tr(const float* __restrict__ src, u16* __restrict__ dst, int R, int C) {
    int i = blockIdx.x * blockDim.x + threadIdx.x;
    if (i >= R * C) return;
    int r = i / C, c = i % C;
    dst[(size_t)c * R + r] = f2bf(src[i]);
}

// ---------------------------------------------------------------- QKV GEMM
// X(16448x768 bf16) @ W^T  -> Q,K bf16 (b,h,s,hd);  V -> f16 transposed (b,h,hd,key[pad 288])
// B tile (64 cols x 32 k, 4KB) staged in LDS via async copy, double buffered.
// K loop fully unrolled: accumulators stay in fixed VGPRs (no phi copies / hazard nops).
__global__ void k_qkv(const u16* __restrict__ X,
                      const u16* __restrict__ WqT, const u16* __restrict__ WkT,
                      const u16* __restrict__ WvT,
                      const float* __restrict__ bq, const float* __restrict__ bk,
                      const float* __restrict__ bv,
                      u16* __restrict__ Qb, u16* __restrict__ Kb, _Float16* __restrict__ Vt) {
    __shared__ __align__(16) u16 sB[2][64 * 32];

    int tid = threadIdx.x;
    int lane = tid & 31, w = tid >> 5;
    int l15 = lane & 15, kb = (lane >> 4) * 8;
    int z = blockIdx.z;
    const u16*   WT   = (z == 0) ? WqT : (z == 1) ? WkT : WvT;
    const float* bias = (z == 0) ? bq  : (z == 1) ? bk  : bv;

    int m0 = blockIdx.x * 128 + w * 16;
    int n0 = blockIdx.y * 64;
    int arow = m0 + l15; if (arow > ROWS - 1) arow = ROWS - 1;

    // cooperative B-tile staging: 256 threads x 16B = 4KB tile
    int lc = tid >> 2;            // col 0..63
    int ch = (tid & 3) * 8;       // 8-element chunk within 32-k band
    const u16* wrow = WT + (size_t)(n0 + lc) * DQ + ch;

    f32x8 c[4] = {zero8(), zero8(), zero8(), zero8()};
    const u16* xrow = X + (size_t)arow * DQ + kb;

    async_ld_b128(wrow, &sB[0][lc * 32 + ch]);          // preload kk=0
#pragma unroll
    for (int it = 0; it < DQ / 32; ++it) {
        int cur = it & 1;
        wait_async0();
        __syncthreads();                                // tile[cur] visible to all waves
        if (it + 1 < DQ / 32)
            async_ld_b128(wrow + (it + 1) * 32, &sB[cur ^ 1][lc * 32 + ch]);

        int kk = it * 32;
        bf16x16 a = ldbf(xrow + kk);
        __builtin_prefetch(xrow + kk + 64, 0, 1);
#pragma unroll
        for (int j = 0; j < 4; ++j)
            c[j] = wmma_bf(a, ldbf(&sB[cur][(j * 16 + l15) * 32 + kb]), c[j]);
    }
#pragma unroll
    for (int j = 0; j < 4; ++j) {
        int n = n0 + j * 16 + l15;
        float bval = bias[n];
#pragma unroll
        for (int r = 0; r < 8; ++r) {
            int m = m0 + kb + r;                  // kb == (lane>>4)*8 row offset
            if (m >= ROWS) continue;
            float val = c[j][r] + bval;
            unsigned bb = (unsigned)m / (unsigned)SQ;
            unsigned s  = (unsigned)m % (unsigned)SQ;
            if (z == 2) {
                int hd = j * 16 + l15;            // n0 is a multiple of 64
                Vt[((size_t)(bb * HQ + blockIdx.y) * HDQ + hd) * SPAD + s] = (_Float16)val;
            } else {
                int h = n >> 6, hd = n & 63;
                u16* dst = (z == 0) ? Qb : Kb;
                dst[((size_t)(bb * HQ + h) * SQ + s) * HDQ + hd] = f2bf(val);
            }
        }
    }
}

// ---------------------------------------------------------------- adversarial branch
__global__ void k_adv(const u16* __restrict__ Kb,
                      const u16* __restrict__ A1T, const float* __restrict__ a1,
                      const float* __restrict__ A2, const float* __restrict__ a2,
                      const u16* __restrict__ B1T, const float* __restrict__ b1,
                      const float* __restrict__ B2, const float* __restrict__ b2,
                      float* __restrict__ Eb, float* __restrict__ lossAcc) {
    __shared__ float p1s[256];
    __shared__ float p2s[64];
    __shared__ float lred[256];

    int tid = threadIdx.x, lane = tid & 31, w = tid >> 5;
    int l15 = lane & 15, kb = (lane >> 4) * 8;
    int bh = blockIdx.x;
    const u16* Kbase = Kb + (size_t)bh * SQ * HDQ;
    float lloc = 0.f;

    // ---- p1: 256 tokens, feat 64 -> hidden 512 -> sigmoid
    for (int mi = 0; mi < 2; ++mi) {
        int mt = w * 2 + mi;
        const u16* arow = Kbase + (size_t)(1 + mt * 16 + l15) * HDQ + kb;
        bf16x16 af0 = ldbf(arow);
        bf16x16 af1 = ldbf(arow + 32);
        float tp[8] = {0.f,0.f,0.f,0.f,0.f,0.f,0.f,0.f};
        for (int nt = 0; nt < 32; ++nt) {
            int col = nt * 16 + l15;
            const u16* brow = A1T + (size_t)col * HDQ + kb;
            f32x8 c = zero8();
            c = wmma_bf(af0, ldbf(brow), c);
            c = wmma_bf(af1, ldbf(brow + 32), c);
            float hb = a1[col], a2v = A2[col];
#pragma unroll
            for (int r = 0; r < 8; ++r) {
                float hv = c[r] + hb;
                if (hv > 0.f) tp[r] += hv * a2v;
            }
        }
        for (int off = 8; off > 0; off >>= 1)
#pragma unroll
            for (int r = 0; r < 8; ++r) tp[r] += __shfl_xor(tp[r], off, 32);
        if (l15 == 0) {
#pragma unroll
            for (int r = 0; r < 8; ++r) {
                int tok = mt * 16 + kb + r;
                float p = 1.f / (1.f + __expf(-(tp[r] + a2[0])));
                p1s[tok] = p;
                lloc += -__logf(fmaxf(1.f - p, EPSQ)) * (1.f / 196608.f);
            }
        }
    }

    // ---- p2: 64 merged tokens, feat 256 -> hidden 512 -> sigmoid
    if (w < 4) {
        int mt = w;
        int t = mt * 16 + l15;
        int gi = t >> 3, gj = t & 7;
        bf16x16 af[8];
#pragma unroll
        for (int ks = 0; ks < 8; ++ks) {
            int f0 = ks * 32 + kb, f1 = f0 + 16;
            int c0 = f0 >> 6, c1 = f1 >> 6;
            int n0 = (2 * gi + (c0 >> 1)) * 16 + (2 * gj + (c0 & 1));
            int n1 = (2 * gi + (c1 >> 1)) * 16 + (2 * gj + (c1 & 1));
            af[ks] = ldbf2(Kbase + (size_t)(1 + n0) * HDQ + (f0 & 63),
                           Kbase + (size_t)(1 + n1) * HDQ + (f1 & 63));
        }
        float tp[8] = {0.f,0.f,0.f,0.f,0.f,0.f,0.f,0.f};
        for (int nt = 0; nt < 32; ++nt) {
            int col = nt * 16 + l15;
            const u16* brow = B1T + (size_t)col * 256 + kb;
            f32x8 c = zero8();
#pragma unroll
            for (int ks = 0; ks < 8; ++ks) c = wmma_bf(af[ks], ldbf(brow + ks * 32), c);
            float hb = b1[col], b2v = B2[col];
#pragma unroll
            for (int r = 0; r < 8; ++r) {
                float hv = c[r] + hb;
                if (hv > 0.f) tp[r] += hv * b2v;
            }
        }
        for (int off = 8; off > 0; off >>= 1)
#pragma unroll
            for (int r = 0; r < 8; ++r) tp[r] += __shfl_xor(tp[r], off, 32);
        if (l15 == 0) {
#pragma unroll
            for (int r = 0; r < 8; ++r) {
                int tok = mt * 16 + kb + r;
                float p = 1.f / (1.f + __expf(-(tp[r] + b2[0])));
                p2s[tok] = p;
                lloc += -__logf(fmaxf(1.f - p, EPSQ)) * (1.f / 49152.f);
            }
        }
    }
    __syncthreads();

    // ---- entropy factors
    {
        int n = tid;                               // 0..255 patch tokens
        int row = n >> 4, col = n & 15;
        float a = 0.525f * p1s[n] + 0.475f * p2s[(row >> 1) * 8 + (col >> 1)];
        float E = -a * __log2f(a + EPSQ) - (1.f - a) * __log2f(1.f - a + EPSQ);
        Eb[(size_t)bh * SQ + 1 + n] = E;
        if (n == 0) Eb[(size_t)bh * SQ] = 1.f;
    }

    // ---- block loss reduction
    lred[tid] = lloc;
    __syncthreads();
    for (int s = 128; s > 0; s >>= 1) {
        if (tid < s) lred[tid] += lred[tid + s];
        __syncthreads();
    }
    if (tid == 0) atomicAdd(lossAcc, lred[0]);
}

// ---------------------------------------------------------------- attention
// per-wave LDS slice: scores f32[16][288] + probs f16[16][288] + inv[16]
#define ATT_SS 18432
#define ATT_SP 9216
#define ATT_PERW (ATT_SS + ATT_SP + 64)   // 27712 bytes, 16B aligned

__global__ void k_attn(const u16* __restrict__ Qb, const u16* __restrict__ Kb,
                       const _Float16* __restrict__ Vt, const float* __restrict__ Eb,
                       u16* __restrict__ ctxb) {
    extern __shared__ char smem[];
    int tid = threadIdx.x, lane = tid & 31, w = tid >> 5;
    int l15 = lane & 15, kb = (lane >> 4) * 8;
    int bh = blockIdx.x;
    int q0 = blockIdx.y * 128 + w * 16;
    if (q0 > 256) return;                         // wave-uniform

    const u16*      Qbh = Qb + (size_t)bh * SQ * HDQ;
    const u16*      Kbh = Kb + (size_t)bh * SQ * HDQ;
    const _Float16* Vbh = Vt + (size_t)bh * HDQ * SPAD;
    float*    sS   = (float*)   (smem + (size_t)w * ATT_PERW);
    _Float16* sP   = (_Float16*)(smem + (size_t)w * ATT_PERW + ATT_SS);
    float*    sInv = (float*)   (smem + (size_t)w * ATT_PERW + ATT_SS + ATT_SP);

    // pass 1: scores tile (16 queries x 272 keys) via bf16 WMMA
    int arow = q0 + l15; if (arow > 256) arow = 256;
    const u16* aptr = Qbh + (size_t)arow * HDQ + kb;
    bf16x16 a0 = ldbf(aptr), a1f = ldbf(aptr + 32);
    for (int nt = 0; nt < 17; ++nt) {
        int key = nt * 16 + l15;
        int keyc = key > 256 ? 256 : key;
        const u16* bptr = Kbh + (size_t)keyc * HDQ + kb;
        f32x8 c = zero8();
        c = wmma_bf(a0,  ldbf(bptr),      c);
        c = wmma_bf(a1f, ldbf(bptr + 32), c);
#pragma unroll
        for (int r = 0; r < 8; ++r)
            sS[(kb + r) * SPAD + nt * 16 + l15] = (key < 257) ? c[r] * 0.125f : -1e30f;
    }

    // pass 2: softmax (entropy folded into row 0, 1/sum deferred to pass 3)
    int row = l15;
    int base = row * SPAD, st = (lane >> 4) * 144;
    float mx = -1e30f;
    for (int i = 0; i < 144; ++i) {
        int idx = st + i;
        if (idx < 257) mx = fmaxf(mx, sS[base + idx]);
    }
    mx = fmaxf(mx, __shfl_xor(mx, 16, 32));
    float sum = 0.f;
    bool erow = (q0 == 0 && row == 0);
    for (int i = 0; i < 144; ++i) {
        int idx = st + i;
        float pv = 0.f;
        if (idx < 257) {
            float e = __expf(sS[base + idx] - mx);
            sum += e;
            float f = erow ? Eb[(size_t)bh * SQ + idx] : 1.f;
            pv = e * f;
        }
        sP[base + idx] = (_Float16)pv;
    }
    sum += __shfl_xor(sum, 16, 32);
    if (lane < 16) sInv[row] = 1.f / sum;

    // pass 3: ctx = probs @ V via f16 WMMA (V pre-transposed + zero-padded)
    f16x16 pa[9];
    const _Float16* prow = sP + (size_t)l15 * SPAD + kb;
#pragma unroll
    for (int ks = 0; ks < 9; ++ks) pa[ks] = ldh(prow + ks * 32);

    int b_ = bh / HQ, h_ = bh % HQ;
#pragma unroll
    for (int j = 0; j < 4; ++j) {
        int hd = j * 16 + l15;
        const _Float16* vcol = Vbh + (size_t)hd * SPAD + kb;
        f32x8 c = zero8();
#pragma unroll
        for (int ks = 0; ks < 9; ++ks) c = wmma_h(pa[ks], ldh(vcol + ks * 32), c);
#pragma unroll
        for (int r = 0; r < 8; ++r) {
            int m = kb + r, q = q0 + m;
            if (q < 257) {
                float val = c[r] * sInv[m];
                ctxb[((size_t)(b_ * SQ + q)) * DQ + h_ * HDQ + hd] = f2bf(val);
            }
        }
    }
}

// ---------------------------------------------------------------- output projection
// same async-LDS-staged B tile scheme as k_qkv, K loop fully unrolled
__global__ void k_out(const u16* __restrict__ X, const u16* __restrict__ WoT,
                      const float* __restrict__ bo, float* __restrict__ out) {
    __shared__ __align__(16) u16 sB[2][64 * 32];

    int tid = threadIdx.x;
    int lane = tid & 31, w = tid >> 5;
    int l15 = lane & 15, kb = (lane >> 4) * 8;
    int m0 = blockIdx.x * 128 + w * 16;
    int n0 = blockIdx.y * 64;
    int arow = m0 + l15; if (arow > ROWS - 1) arow = ROWS - 1;

    int lc = tid >> 2;
    int ch = (tid & 3) * 8;
    const u16* wrow = WoT + (size_t)(n0 + lc) * DQ + ch;

    f32x8 c[4] = {zero8(), zero8(), zero8(), zero8()};
    const u16* xrow = X + (size_t)arow * DQ + kb;

    async_ld_b128(wrow, &sB[0][lc * 32 + ch]);
#pragma unroll
    for (int it = 0; it < DQ / 32; ++it) {
        int cur = it & 1;
        wait_async0();
        __syncthreads();
        if (it + 1 < DQ / 32)
            async_ld_b128(wrow + (it + 1) * 32, &sB[cur ^ 1][lc * 32 + ch]);

        int kk = it * 32;
        bf16x16 a = ldbf(xrow + kk);
        __builtin_prefetch(xrow + kk + 64, 0, 1);
#pragma unroll
        for (int j = 0; j < 4; ++j)
            c[j] = wmma_bf(a, ldbf(&sB[cur][(j * 16 + l15) * 32 + kb]), c[j]);
    }
#pragma unroll
    for (int j = 0; j < 4; ++j) {
        int n = n0 + j * 16 + l15;
        float bval = bo[n];
#pragma unroll
        for (int r = 0; r < 8; ++r) {
            int m = m0 + kb + r;
            if (m < ROWS) out[(size_t)m * DQ + n] = c[j][r] + bval;
        }
    }
}

__global__ void k_fin(const float* __restrict__ acc, float* __restrict__ dst) {
    dst[0] = acc[0];
}

// ---------------------------------------------------------------- launch

extern "C" void kernel_launch(void* const* d_in, const int* in_sizes, int n_in,
                              void* d_out, int out_size, void* d_ws, size_t ws_size,
                              hipStream_t stream) {
    const float* hidden = (const float*)d_in[0];
    const float* Wq = (const float*)d_in[1];  const float* bq = (const float*)d_in[2];
    const float* Wk = (const float*)d_in[3];  const float* bk = (const float*)d_in[4];
    const float* Wv = (const float*)d_in[5];  const float* bv = (const float*)d_in[6];
    const float* Wo = (const float*)d_in[7];  const float* bo = (const float*)d_in[8];
    const float* A1 = (const float*)d_in[9];  const float* a1 = (const float*)d_in[10];
    const float* A2 = (const float*)d_in[11]; const float* a2 = (const float*)d_in[12];
    const float* B1 = (const float*)d_in[13]; const float* b1 = (const float*)d_in[14];
    const float* B2 = (const float*)d_in[15]; const float* b2 = (const float*)d_in[16];
    (void)in_sizes; (void)n_in; (void)out_size; (void)ws_size;

    char* ws = (char*)d_ws;
    size_t off = 0;
    auto alc = [&](size_t bytes) { size_t o = off; off = (off + bytes + 255) & ~(size_t)255; return o; };

    const size_t nHid = (size_t)ROWS * DQ;                // 12,632,064
    u16*      hidbf = (u16*)     (ws + alc(nHid * 2));
    u16*      WqT   = (u16*)     (ws + alc((size_t)DQ * DQ * 2));
    u16*      WkT   = (u16*)     (ws + alc((size_t)DQ * DQ * 2));
    u16*      WvT   = (u16*)     (ws + alc((size_t)DQ * DQ * 2));
    u16*      WoT   = (u16*)     (ws + alc((size_t)DQ * DQ * 2));
    u16*      A1T   = (u16*)     (ws + alc((size_t)ADH * HDQ * 2));
    u16*      B1T   = (u16*)     (ws + alc((size_t)ADH * 256 * 2));
    u16*      Qbuf  = (u16*)     (ws + alc((size_t)BQ * HQ * SQ * HDQ * 2));
    u16*      Kbuf  = (u16*)     (ws + alc((size_t)BQ * HQ * SQ * HDQ * 2));
    size_t    VtBytes = (size_t)BQ * HQ * HDQ * SPAD * 2;
    _Float16* Vt    = (_Float16*)(ws + alc(VtBytes));
    float*    Ebuf  = (float*)   (ws + alc((size_t)BQ * HQ * SQ * 4));
    u16*      ctxb  = (u16*)     (ws + alc(nHid * 2));
    float*    lossA = (float*)   (ws + alc(256));

    hipMemsetAsync(Vt, 0, VtBytes, stream);       // zero pad keys 257..287
    hipMemsetAsync(lossA, 0, 4, stream);

    // prep: convert + transpose
    k_cvt<<<dim3((unsigned)(nHid / 256)), dim3(256), 0, stream>>>(hidden, hidbf, (int)nHid);
    k_tr <<<dim3(2304), dim3(256), 0, stream>>>(Wq, WqT, DQ, DQ);
    k_tr <<<dim3(2304), dim3(256), 0, stream>>>(Wk, WkT, DQ, DQ);
    k_tr <<<dim3(2304), dim3(256), 0, stream>>>(Wv, WvT, DQ, DQ);
    k_tr <<<dim3(2304), dim3(256), 0, stream>>>(Wo, WoT, DQ, DQ);
    k_tr <<<dim3(128),  dim3(256), 0, stream>>>(A1, A1T, HDQ, ADH);
    k_tr <<<dim3(512),  dim3(256), 0, stream>>>(B1, B1T, 256, ADH);

    // QKV projections (WMMA bf16, async-LDS weight staging)
    k_qkv<<<dim3(129, 12, 3), dim3(256), 0, stream>>>(hidbf, WqT, WkT, WvT,
                                                      bq, bk, bv, Qbuf, Kbuf, Vt);

    // adversarial branch + entropy + loss (WMMA bf16)
    k_adv<<<dim3(BQ * HQ), dim3(256), 0, stream>>>(Kbuf, A1T, a1, A2, a2,
                                                   B1T, b1, B2, b2, Ebuf, lossA);

    // attention: scores/softmax/ctx (WMMA bf16 + f16)
    k_attn<<<dim3(BQ * HQ, 3), dim3(256), (size_t)8 * ATT_PERW, stream>>>(
        Qbuf, Kbuf, Vt, Ebuf, ctxb);

    // output projection -> d_out (fp32)
    k_out<<<dim3(129, 12), dim3(256), 0, stream>>>(ctxb, WoT, bo, (float*)d_out);

    // scalar loss
    k_fin<<<dim3(1), dim3(1), 0, stream>>>(lossA, (float*)d_out + (size_t)ROWS * DQ);
}
